// SpatialCrossAttention_16552803958966
// MI455X (gfx1250) — matrix operand order repaired and checked by reading, not compiled, not run on hardware
//
#include <hip/hip_runtime.h>

typedef __attribute__((ext_vector_type(16))) _Float16 v16h;
typedef __attribute__((ext_vector_type(8)))  _Float16 v8h;
typedef __attribute__((ext_vector_type(8)))  float    v8f;
typedef __attribute__((ext_vector_type(4)))  float    v4f;

#define BATCH 8
#define CCH   256
#define NSP   4096
#define QROW  32            // q/k row storage: 16 real + 16 zero-pad halfs
#define NRT   34            // 544/16 output row tiles in fused projection
#define ALD_STRIDE 4104     // 4096 + 8 halfs pad (bank spread), 16B-aligned rows

// D = A(16x32 f16) * B(32x16 f16) + C(16x16 f32)
__device__ __forceinline__ v8f wmma_f16(v16h a, v16h b, v8f c) {
  return __builtin_amdgcn_wmma_f32_16x16x32_f16(false, a, false, b, (short)0, c,
                                                false, false);
}

// A-operand: lane M = lane&15; elems 0..7 -> K = k0 + 8*hsel + e,
//            elems 8..15 -> K = k0 + 16 + 8*hsel + (e-8)   (hsel = lane>=16)
__device__ __forceinline__ v16h ldA(const _Float16* __restrict__ row, int k0, int hsel) {
  v16h a;
  ((v8h*)&a)[0] = *(const v8h*)(row + k0 + 8 * hsel);
  ((v8h*)&a)[1] = *(const v8h*)(row + k0 + 16 + 8 * hsel);
  return a;
}
// B-operand: lane N = lane&15; elems 0..7 -> K = k0 + 8*hsel + e,
//            elems 8..15 -> K = k0 + 16 + 8*hsel + (e-8)
__device__ __forceinline__ v16h ldB(const _Float16* __restrict__ col, int k0, int hsel) {
  v16h b;
  ((v8h*)&b)[0] = *(const v8h*)(col + k0 + 8 * hsel);
  ((v8h*)&b)[1] = *(const v8h*)(col + k0 + 16 + 8 * hsel);
  return b;
}

// ---------------- Kernel 0: pack fused f16 weight matrix [544][512] ----------------
// rows 0-15: Wq | 16-31: Wk | 32-287: [Wv1 | 0] | 288-543: [0 | Wv2]
__global__ void k_buildw(const float* __restrict__ Wq, const float* __restrict__ bq,
                         const float* __restrict__ Wk, const float* __restrict__ bk,
                         const float* __restrict__ Wv1, const float* __restrict__ bv1,
                         const float* __restrict__ Wv2, const float* __restrict__ bv2,
                         _Float16* __restrict__ Wbig, float* __restrict__ bbig) {
  int idx = blockIdx.x * blockDim.x + threadIdx.x;
  if (idx < 544 * 512) {
    int r = idx >> 9, c = idx & 511;
    float w;
    if (r < 16)       w = Wq[r * 512 + c];
    else if (r < 32)  w = Wk[(r - 16) * 512 + c];
    else if (r < 288) w = (c < 256)  ? Wv1[(r - 32) * 256 + c] : 0.f;
    else              w = (c >= 256) ? Wv2[(r - 288) * 256 + (c - 256)] : 0.f;
    Wbig[idx] = (_Float16)w;
  }
  if (idx < 544) {
    float bb;
    if (idx < 16)       bb = bq[idx];
    else if (idx < 32)  bb = bk[idx - 16];
    else if (idx < 288) bb = bv1[idx - 32];
    else                bb = bv2[idx - 288];
    bbig[idx] = bb;
  }
}

// ---------------- Kernel 1: fused q/k/v projections via WMMA ----------------
// grid (N/64, B), 256 threads, dyn LDS = 64*520*2 bytes (x^T tile in f16)
__global__ void k_qkv(const float* __restrict__ x1, const float* __restrict__ x2,
                      const _Float16* __restrict__ Wbig, const float* __restrict__ bbig,
                      _Float16* __restrict__ qh, _Float16* __restrict__ kh,
                      _Float16* __restrict__ v1h, _Float16* __restrict__ v2h) {
  extern __shared__ char smem[];
  _Float16* xT = (_Float16*)smem;                 // [64 n][520 c-pad]
  const int b = blockIdx.y, n0 = blockIdx.x * 64;
  const int tid = threadIdx.x, lane = tid & 31;
  const int wave = __builtin_amdgcn_readfirstlane(tid >> 5);  // SGPR-uniform
  const int hsel = lane >> 4;

  // stage x^T tile (concat channel dim) as f16, coalesced global reads
  for (int idx = tid; idx < 512 * 64; idx += 256) {
    int c = idx >> 6, j = idx & 63;
    float xv = (c < 256) ? x1[(size_t)(b * 256 + c) * NSP + n0 + j]
                         : x2[(size_t)(b * 256 + c - 256) * NSP + n0 + j];
    xT[j * 520 + c] = (_Float16)xv;
  }
  __syncthreads();

  for (int t = wave; t < NRT * 4; t += 8) {       // 17 tiles per wave, scalar loop
    int rt = t >> 2, nt = t & 3;
    int kks = (rt >= 18) ? 8 : 0;                 // skip known-zero K blocks
    int kke = (rt < 2) ? 16 : kks + 8;
    const _Float16* wrow = Wbig + (size_t)(rt * 16 + (lane & 15)) * 512;
    const _Float16* xrow = xT + (nt * 16 + (lane & 15)) * 520;
    v8f acc = {};
    for (int kk = kks; kk < kke; ++kk)
      acc = wmma_f16(ldA(wrow, kk * 32, hsel), ldB(xrow, kk * 32, hsel), acc);

    int n = n0 + nt * 16 + (lane & 15);
    if (rt < 2) {                                 // q or k rows: store [n][d] padded
      _Float16* dst = (rt == 0 ? qh : kh) + ((size_t)b * NSP + n) * QROW;
      v8h pack;
#pragma unroll
      for (int e = 0; e < 8; ++e)
        pack[e] = (_Float16)(acc[e] + bbig[rt * 16 + e + 8 * hsel]);
      *(v8h*)(dst + 8 * hsel) = pack;
      v8h z = {};
      *(v8h*)(dst + 16) = z;                      // zero K-pad (dup writes benign)
      *(v8h*)(dst + 24) = z;
    } else {                                      // v1/v2 rows: store [c][n]
      bool isv1 = rt < 18;
      _Float16* dst = isv1 ? v1h : v2h;
      int cbase = (isv1 ? (rt - 2) : (rt - 18)) * 16 + 8 * hsel;
#pragma unroll
      for (int e = 0; e < 8; ++e)
        dst[((size_t)b * CCH + cbase + e) * NSP + n] =
            (_Float16)(acc[e] + bbig[rt * 16 + e + 8 * hsel]);
    }
  }
}

// ---------------- Kernel 2: fused energy + softmax + V@A^T (+residual) ----------------
// grid (N/32, B), 512 threads (16 waves), dyn LDS = 32*ALD_STRIDE*2 = 262656 B
__global__ void k_attn(const _Float16* __restrict__ qh, const _Float16* __restrict__ kh,
                       const _Float16* __restrict__ v1h, const _Float16* __restrict__ v2h,
                       const float* __restrict__ x1, const float* __restrict__ x2,
                       float* __restrict__ out) {
  extern __shared__ char smem[];
  _Float16* Alds = (_Float16*)smem;               // [32 n][ALD_STRIDE m]
  const int b = blockIdx.y, n0 = blockIdx.x * 32;
  const int tid = threadIdx.x, lane = tid & 31;
  const int wave = __builtin_amdgcn_readfirstlane(tid >> 5);  // SGPR-uniform
  const int hsel = lane >> 4;
  const size_t OUT2 = (size_t)BATCH * CCH * NSP;  // out2 offset (floats)
  const size_t ATTN = 2 * OUT2;                   // attention offset (floats)

  // Pass 1: energy tiles (K=16 zero-padded to 32) -> LDS f16
  {
    int r = wave >> 3;                            // row tile 0/1 (8 waves each)
    const _Float16* qrow = qh + ((size_t)b * NSP + n0 + 16 * r + (lane & 15)) * QROW;
    v16h aQ = ldA(qrow, 0, hsel);
    for (int mt = (wave & 7); mt < 256; mt += 8) {
      int m0 = mt * 16;
      const _Float16* kcol = kh + ((size_t)b * NSP + m0 + (lane & 15)) * QROW;
      v8f e = {};
      e = wmma_f16(aQ, ldB(kcol, 0, hsel), e);
#pragma unroll
      for (int v = 0; v < 8; ++v)
        Alds[(16 * r + v + 8 * hsel) * ALD_STRIDE + m0 + (lane & 15)] = (_Float16)e[v];
    }
  }
  __syncthreads();

  // Pass 2/3: per-row softmax, vectorized v8h; one wave owns 2 rows
  for (int rr = 0; rr < 2; ++rr) {
    int n = wave * 2 + rr;
    _Float16* rowp = Alds + n * ALD_STRIDE;
    float mx = -3.402823466e38f;
    for (int it = 0; it < 16; ++it) {
      v8h h = *(const v8h*)(rowp + it * 256 + lane * 8);
#pragma unroll
      for (int e = 0; e < 8; ++e) mx = fmaxf(mx, (float)h[e]);
    }
#pragma unroll
    for (int off = 16; off; off >>= 1) mx = fmaxf(mx, __shfl_xor(mx, off, 32));
    float s = 0.f;
    for (int it = 0; it < 16; ++it) {
      _Float16* p8 = rowp + it * 256 + lane * 8;
      v8h h = *(const v8h*)p8, o;
#pragma unroll
      for (int e = 0; e < 8; ++e) {
        float p = __expf((float)h[e] - mx);
        s += p;
        o[e] = (_Float16)p;
      }
      *(v8h*)p8 = o;
    }
#pragma unroll
    for (int off = 16; off; off >>= 1) s += __shfl_xor(s, off, 32);
    float inv = 1.f / s;
    float* att = out + ATTN + ((size_t)b * NSP + n0 + n) * NSP;
    for (int it = 0; it < 16; ++it) {
      int m = it * 256 + lane * 8;
      _Float16* p8 = rowp + m;
      v8h h = *(const v8h*)p8, o;
      v4f f0, f1;
#pragma unroll
      for (int e = 0; e < 8; ++e) {
        float p = (float)h[e] * inv;
        o[e] = (_Float16)p;                       // normalized prob for pass 4
        if (e < 4) f0[e] = p; else f1[e - 4] = p;
      }
      *(v8h*)p8 = o;
      __builtin_nontemporal_store(f0, (v4f*)(att + m));      // streaming fp32 out
      __builtin_nontemporal_store(f1, (v4f*)(att + m + 4));
    }
  }
  __syncthreads();

  // Pass 4: out[c,n] = sum_m V[c,m] * A[n,m] (+residual); dual n-tiles share V load
  for (int tp = wave; tp < 32; tp += 16) {        // 2 (mat,ct) pairs per wave
    int mat = tp >> 4, ct = tp & 15;
    const _Float16* V = (mat ? v2h : v1h) + (size_t)b * CCH * NSP;
    const float* X    = (mat ? x2 : x1) + (size_t)b * CCH * NSP;
    float* O          = out + (mat ? OUT2 : 0) + (size_t)b * CCH * NSP;
    const _Float16* vrow  = V + (size_t)(ct * 16 + (lane & 15)) * NSP;
    const _Float16* arow0 = Alds + (lane & 15) * ALD_STRIDE;         // n-tile 0
    const _Float16* arow1 = Alds + (16 + (lane & 15)) * ALD_STRIDE;  // n-tile 1
    v8f acc0 = {}, acc1 = {};
#pragma unroll 4
    for (int mt = 0; mt < 128; ++mt) {
      int m0 = mt * 32;
      v16h aV = ldA(vrow, m0, hsel);              // shared A-operand
      acc0 = wmma_f16(aV, ldB(arow0, m0, hsel), acc0);
      acc1 = wmma_f16(aV, ldB(arow1, m0, hsel), acc1);
    }
    int nn = n0 + (lane & 15);
#pragma unroll
    for (int v = 0; v < 8; ++v) {
      size_t idx0 = (size_t)(ct * 16 + v + 8 * hsel) * NSP + nn;
      __builtin_nontemporal_store(acc0[v] + X[idx0], O + idx0);
      __builtin_nontemporal_store(acc1[v] + X[idx0 + 16], O + idx0 + 16);
    }
  }
}

extern "C" void kernel_launch(void* const* d_in, const int* in_sizes, int n_in,
                              void* d_out, int out_size, void* d_ws, size_t ws_size,
                              hipStream_t stream) {
  (void)in_sizes; (void)n_in; (void)out_size; (void)ws_size;
  const float* x1  = (const float*)d_in[0];
  const float* x2  = (const float*)d_in[1];
  const float* Wq  = (const float*)d_in[2];
  const float* bq  = (const float*)d_in[3];
  const float* Wk  = (const float*)d_in[4];
  const float* bk  = (const float*)d_in[5];
  const float* Wv1 = (const float*)d_in[6];
  const float* bv1 = (const float*)d_in[7];
  const float* Wv2 = (const float*)d_in[8];
  const float* bv2 = (const float*)d_in[9];

  char* ws = (char*)d_ws;                          // ~39 MB used
  _Float16* Wbig = (_Float16*)(ws);                // 544*512*2 = 544 KB
  float*    bbig = (float*)(ws + 0x90000);         // 544 floats
  _Float16* qh   = (_Float16*)(ws + 0x100000);     // [B][N][32] f16, 2 MB
  _Float16* kh   = (_Float16*)(ws + 0x300000);     // [B][N][32] f16, 2 MB
  _Float16* v1h  = (_Float16*)(ws + 0x500000);     // [B][C][N] f16, 16 MB
  _Float16* v2h  = (_Float16*)(ws + 0x1500000);    // [B][C][N] f16, 16 MB

  k_buildw<<<1088, 256, 0, stream>>>(Wq, bq, Wk, bk, Wv1, bv1, Wv2, bv2, Wbig, bbig);
  k_qkv<<<dim3(64, BATCH), 256, 64 * 520 * 2, stream>>>(x1, x2, Wbig, bbig, qh, kh, v1h, v2h);
  k_attn<<<dim3(128, BATCH), 512, 32 * ALD_STRIDE * 2, stream>>>(qh, kh, v1h, v2h, x1, x2,
                                                                 (float*)d_out);
}
